// MultiheadAttention_83021717832256
// MI455X (gfx1250) — matrix operand order repaired and checked by reading, not compiled, not run on hardware
//
#include <hip/hip_runtime.h>

// MI455X / gfx1250, wave32. All matmuls via v_wmma_f32_16x16x32_bf16.
typedef __attribute__((ext_vector_type(16))) __bf16 v16bf;
typedef __attribute__((ext_vector_type(8)))  __bf16 v8bf;
typedef __attribute__((ext_vector_type(8)))  float  v8f;

#define T_LEN 2048
#define S_LEN 2048
#define BSZ   4
#define EMB   512
#define NH    8
#define HD    64

// ---- wave32 16-lane butterfly reductions (masks <16 stay inside each half) ----
__device__ __forceinline__ float xor16_max(float v) {
  #pragma unroll
  for (int m = 1; m < 16; m <<= 1) v = fmaxf(v, __shfl_xor(v, m, 32));
  return v;
}
__device__ __forceinline__ float xor16_sum(float v) {
  #pragma unroll
  for (int m = 1; m < 16; m <<= 1) v += __shfl_xor(v, m, 32);
  return v;
}

// =====================================================================
// One-shot f32 -> bf16 conversion (vectorized 8/thread). Removes all
// redundant per-tile conversions from the GEMM hot loops.
// =====================================================================
__global__ __launch_bounds__(256)
void cvt_kernel(const float* __restrict__ src, __bf16* __restrict__ dst, int n8)
{
  const int i = blockIdx.x * 256 + threadIdx.x;
  if (i < n8) {
    v8f f = *(const v8f*)(src + (size_t)i * 8);
    v8bf o;
    #pragma unroll
    for (int j = 0; j < 8; ++j) o[j] = (__bf16)f[j];
    *(v8bf*)(dst + (size_t)i * 8) = o;
  }
}

// =====================================================================
// Projection GEMM: y = x @ W^T + b. x_bf:[M=8192,K=512], W_bf:[N=512,K=512],
// both pre-converted bf16. One wave per 16x64 output tile: per k-step
// 1 A-frag load (reused 4x) + 4 B-frag loads + 4 independent WMMAs.
// mode 0: Q -> bf16 [b,h,t,d] * hd^-0.5
// mode 1: K -> bf16 [b,h,s,d]
// mode 2: V -> bf16 [b,h,d,s]   (transposed for PV B-matrix loads)
// mode 3: out-proj -> f32 [T,B,E] straight to d_out
// =====================================================================
__global__ __launch_bounds__(128)
void proj_kernel(const __bf16* __restrict__ x_bf, const __bf16* __restrict__ W_bf,
                 const float* __restrict__ bias, int mode,
                 __bf16* __restrict__ qkv_bf, float* __restrict__ out_f32)
{
  const int lane   = threadIdx.x & 31;
  const int wave   = threadIdx.x >> 5;
  const int tile   = blockIdx.x * 4 + wave;   // 4096 tiles (512 M x 8 N64)
  const int tile_m = tile >> 3;               // 512 M-tiles of 16
  const int tile_n = tile & 7;                // 8 N-tiles of 64
  const int mrow   = lane & 15;
  const int hi     = lane >> 4;

  const __bf16* arow = x_bf + (size_t)(tile_m * 16 + mrow) * EMB;  // A row (lane = M)
  const __bf16* brow = W_bf + (size_t)(tile_n * 64 + mrow) * EMB;  // B col base

  v8f acc[4];
  #pragma unroll
  for (int nt = 0; nt < 4; ++nt) acc[nt] = v8f{};

  for (int kk = 0; kk < EMB; kk += 32) {
    __builtin_prefetch(arow + kk + 128, 0, 1);          // global_prefetch_b8
    // A 16x32: per lane two contiguous 8-elem K chunks at k0=hi*16 and k0+8
    v8bf lo = *(const v8bf*)(arow + kk + hi * 16);
    v8bf up = *(const v8bf*)(arow + kk + hi * 16 + 8);
    v16bf a;
    #pragma unroll
    for (int j = 0; j < 8; ++j) { a[j] = lo[j]; a[8 + j] = up[j]; }
    // 4 B 32x16 frags: per lane 16 contiguous K at k=kk+hi*16 (W is [N,K] row-major)
    #pragma unroll
    for (int nt = 0; nt < 4; ++nt) {
      v16bf bm = *(const v16bf*)(brow + (size_t)nt * 16 * EMB + kk + hi * 16);
      acc[nt] = __builtin_amdgcn_wmma_f32_16x16x32_bf16(false, a, false, bm,
                                                        (short)0, acc[nt], false, false);
    }
  }

  #pragma unroll
  for (int nt = 0; nt < 4; ++nt) {
    const int  ncol = tile_n * 64 + nt * 16 + (lane & 15);
    const float bn  = bias[ncol];
    #pragma unroll
    for (int r = 0; r < 8; ++r) {           // C/D layout: M = r + 8*hi, N = lane&15
      const int Mrow = tile_m * 16 + r + 8 * hi;
      float val = acc[nt][r] + bn;
      if (mode == 3) {
        out_f32[(size_t)Mrow * EMB + ncol] = val;       // [T,B,E] flat == [M,E]
      } else {
        const int t = Mrow >> 2, b = Mrow & 3;          // row = t*BSZ + b
        const int h = ncol >> 6, d = ncol & 63;
        const int bh = b * NH + h;
        if (mode == 0)      qkv_bf[((size_t)bh * T_LEN + t) * HD + d] = (__bf16)(val * 0.125f);
        else if (mode == 1) qkv_bf[((size_t)bh * S_LEN + t) * HD + d] = (__bf16)val;
        else                qkv_bf[((size_t)bh * HD + d) * S_LEN + t] = (__bf16)val;
      }
    }
  }
}

// =====================================================================
// Pass 1 of softmax: per (b,h,t) row -> (max, sum_exp) with online update.
// One wave per (b,h,16-t tile); streams QK^T over all S via WMMA.
// =====================================================================
__global__ __launch_bounds__(128)
void stats_kernel(const __bf16* __restrict__ q_bf, const __bf16* __restrict__ k_bf,
                  const float* __restrict__ mask, float* __restrict__ stats)
{
  const int lane   = threadIdx.x & 31;
  const int wave   = threadIdx.x >> 5;
  const int w      = blockIdx.x * 4 + wave;   // 4096 waves
  const int bh     = w >> 7;
  const int tile_t = w & 127;
  const int mrow   = lane & 15;
  const int hi     = lane >> 4;
  const int t0     = tile_t * 16;

  // Q 16x64 -> two A-matrices (invariant over S)
  const __bf16* qrow = q_bf + ((size_t)bh * T_LEN + t0 + mrow) * HD;
  v16bf aq[2];
  #pragma unroll
  for (int c = 0; c < 2; ++c) {
    v8bf lo = *(const v8bf*)(qrow + c * 32 + hi * 16);
    v8bf up = *(const v8bf*)(qrow + c * 32 + hi * 16 + 8);
    #pragma unroll
    for (int j = 0; j < 8; ++j) { aq[c][j] = lo[j]; aq[c][8 + j] = up[j]; }
  }

  float rm[8], rl[8];
  #pragma unroll
  for (int r = 0; r < 8; ++r) { rm[r] = -3.0e38f; rl[r] = 0.f; }

  for (int s0 = 0; s0 < S_LEN; s0 += 16) {
    const __bf16* krow = k_bf + ((size_t)bh * S_LEN + s0 + mrow) * HD;  // B col = s
    v8f acc = {};
    #pragma unroll
    for (int c = 0; c < 2; ++c) {
      v16bf bk = *(const v16bf*)(krow + c * 32 + hi * 16);
      acc = __builtin_amdgcn_wmma_f32_16x16x32_bf16(false, aq[c], false, bk,
                                                    (short)0, acc, false, false);
    }
    const int scol = s0 + mrow;
    #pragma unroll
    for (int r = 0; r < 8; ++r) {
      const int trow = t0 + r + 8 * hi;
      float scr  = acc[r] + mask[(size_t)trow * S_LEN + scol];
      float tmax = xor16_max(scr);
      float nm   = fmaxf(rm[r], tmax);
      float e    = __expf(scr - nm);
      float es   = xor16_sum(e);
      rl[r] = rl[r] * __expf(rm[r] - nm) + es;
      rm[r] = nm;
    }
  }
  if (mrow == 0) {                    // one lane per 16-group holds the result
    #pragma unroll
    for (int r = 0; r < 8; ++r) {
      const int trow = t0 + r + 8 * hi;
      stats[((size_t)bh * T_LEN + trow) * 2 + 0] = rm[r];
      stats[((size_t)bh * T_LEN + trow) * 2 + 1] = rl[r];
    }
  }
}

// =====================================================================
// Pass 2: probabilities + P@V + head-averaged attn_weights.
// Workgroup = (b, 16-t tile); 8 waves = 8 heads. LDS stages P tiles:
// gives the C-layout -> A-layout transpose AND the cross-head reduction.
// O written as bf16 [T*B, E] feeding the pure-bf16 out-projection GEMM.
// =====================================================================
__global__ __launch_bounds__(256)
void attn_kernel(const __bf16* __restrict__ q_bf, const __bf16* __restrict__ k_bf,
                 const __bf16* __restrict__ v_t, const float* __restrict__ mask,
                 const float* __restrict__ stats, __bf16* __restrict__ o_bf,
                 float* __restrict__ aw)
{
  __shared__ float p_sh[NH][16 * 32];
  const int tid    = threadIdx.x;
  const int lane   = tid & 31;
  const int h      = tid >> 5;              // wave == head
  const int b      = blockIdx.x >> 7;
  const int tile_t = blockIdx.x & 127;
  const int t0     = tile_t * 16;
  const int bh     = b * NH + h;
  const int mrow   = lane & 15;
  const int hi     = lane >> 4;

  const __bf16* qrow = q_bf + ((size_t)bh * T_LEN + t0 + mrow) * HD;
  v16bf aq[2];
  #pragma unroll
  for (int c = 0; c < 2; ++c) {
    v8bf lo = *(const v8bf*)(qrow + c * 32 + hi * 16);
    v8bf up = *(const v8bf*)(qrow + c * 32 + hi * 16 + 8);
    #pragma unroll
    for (int j = 0; j < 8; ++j) { aq[c][j] = lo[j]; aq[c][8 + j] = up[j]; }
  }

  float rm[8], rinv[8];
  #pragma unroll
  for (int r = 0; r < 8; ++r) {
    const int trow = t0 + r + 8 * hi;
    rm[r]    = stats[((size_t)bh * T_LEN + trow) * 2 + 0];
    float l  = stats[((size_t)bh * T_LEN + trow) * 2 + 1];
    rinv[r]  = (l > 0.f) ? 1.f / l : 0.f;
  }

  v8f oacc[4];
  #pragma unroll
  for (int nt = 0; nt < 4; ++nt) oacc[nt] = v8f{};

  for (int s0 = 0; s0 < S_LEN; s0 += 32) {
    #pragma unroll
    for (int st = 0; st < 2; ++st) {
      const int sbase = s0 + st * 16;
      const __bf16* krow = k_bf + ((size_t)bh * S_LEN + sbase + mrow) * HD;
      v8f acc = {};
      #pragma unroll
      for (int c = 0; c < 2; ++c) {
        v16bf bk = *(const v16bf*)(krow + c * 32 + hi * 16);
        acc = __builtin_amdgcn_wmma_f32_16x16x32_bf16(false, aq[c], false, bk,
                                                      (short)0, acc, false, false);
      }
      const int scol = sbase + mrow;
      #pragma unroll
      for (int r = 0; r < 8; ++r) {
        const int trow = t0 + r + 8 * hi;
        float p = __expf(acc[r] + mask[(size_t)trow * S_LEN + scol] - rm[r]) * rinv[r];
        p_sh[h][(r + 8 * hi) * 32 + st * 16 + mrow] = p;
      }
    }
    __syncthreads();
    // attn_weights tile: mean over heads, written once per block
    for (int e = tid; e < 512; e += 256) {
      const int row = e >> 5, col = e & 31;
      float sum = 0.f;
      #pragma unroll
      for (int hh = 0; hh < NH; ++hh) sum += p_sh[hh][e];
      aw[(size_t)(b * T_LEN + t0 + row) * S_LEN + s0 + col] = sum * 0.125f;
    }
    // P tile (16x32) back as bf16 A-matrix for this head
    v16bf ap;
    #pragma unroll
    for (int j = 0; j < 8; ++j) {
      ap[j]     = (__bf16)p_sh[h][mrow * 32 + hi * 16 + j];
      ap[8 + j] = (__bf16)p_sh[h][mrow * 32 + hi * 16 + 8 + j];
    }
    #pragma unroll
    for (int nt = 0; nt < 4; ++nt) {
      const int d = nt * 16 + mrow;                       // B col = d
      const __bf16* vrow = v_t + ((size_t)bh * HD + d) * S_LEN + s0 + hi * 16;
      v16bf bv = *(const v16bf*)vrow;                     // 16 contiguous s
      oacc[nt] = __builtin_amdgcn_wmma_f32_16x16x32_bf16(false, ap, false, bv,
                                                         (short)0, oacc[nt], false, false);
    }
    __syncthreads();
  }

  // O tile 16(t) x 64(d) -> bf16 [T*B, E] for the out-projection
  #pragma unroll
  for (int nt = 0; nt < 4; ++nt) {
    const int e = h * HD + nt * 16 + mrow;
    #pragma unroll
    for (int r = 0; r < 8; ++r) {
      const int t = t0 + r + 8 * hi;
      o_bf[((size_t)t * BSZ + b) * EMB + e] = (__bf16)oacc[nt][r];
    }
  }
}

// =====================================================================
extern "C" void kernel_launch(void* const* d_in, const int* in_sizes, int n_in,
                              void* d_out, int out_size, void* d_ws, size_t ws_size,
                              hipStream_t stream) {
  const float* query = (const float*)d_in[0];
  const float* key   = (const float*)d_in[1];
  const float* value = (const float*)d_in[2];
  const float* amask = (const float*)d_in[3];
  const float* Wq = (const float*)d_in[4];  const float* bq = (const float*)d_in[5];
  const float* Wk = (const float*)d_in[6];  const float* bk = (const float*)d_in[7];
  const float* Wv = (const float*)d_in[8];  const float* bv = (const float*)d_in[9];
  const float* Wo = (const float*)d_in[10]; const float* bo = (const float*)d_in[11];

  char* ws = (char*)d_ws;
  __bf16* wq_bf = (__bf16*)(ws + 0x0000000);   // 512 KB each: W in bf16
  __bf16* wk_bf = (__bf16*)(ws + 0x0080000);
  __bf16* wv_bf = (__bf16*)(ws + 0x0100000);
  __bf16* wo_bf = (__bf16*)(ws + 0x0180000);
  __bf16* xq_bf = (__bf16*)(ws + 0x0200000);   // 8 MB each: activations bf16
  __bf16* xk_bf = (__bf16*)(ws + 0x0A00000);
  __bf16* xv_bf = (__bf16*)(ws + 0x1200000);
  __bf16* q_bf  = (__bf16*)(ws + 0x1A00000);   // 8 MB  [b,h,t,d]
  __bf16* k_bf  = (__bf16*)(ws + 0x2200000);   // 8 MB  [b,h,s,d]
  __bf16* v_t   = (__bf16*)(ws + 0x2A00000);   // 8 MB  [b,h,d,s]
  float*  stats = (float*) (ws + 0x3200000);   // 512 KB (m,l) per row
  __bf16* o_bf  = (__bf16*)(ws + 0x3280000);   // 8 MB  [T*B, E]

  float* out = (float*)d_out;                  // [T,B,E]
  float* aw  = out + (size_t)T_LEN * BSZ * EMB;// [B,T,S]

  const int W8 = (EMB * EMB) / 8;              // 32768 vec8 per weight
  const int X8 = (T_LEN * BSZ * EMB) / 8;      // 524288 vec8 per activation
  cvt_kernel<<<(W8 + 255) / 256, 256, 0, stream>>>(Wq, wq_bf, W8);
  cvt_kernel<<<(W8 + 255) / 256, 256, 0, stream>>>(Wk, wk_bf, W8);
  cvt_kernel<<<(W8 + 255) / 256, 256, 0, stream>>>(Wv, wv_bf, W8);
  cvt_kernel<<<(W8 + 255) / 256, 256, 0, stream>>>(Wo, wo_bf, W8);
  cvt_kernel<<<(X8 + 255) / 256, 256, 0, stream>>>(query, xq_bf, X8);
  cvt_kernel<<<(X8 + 255) / 256, 256, 0, stream>>>(key,   xk_bf, X8);
  cvt_kernel<<<(X8 + 255) / 256, 256, 0, stream>>>(value, xv_bf, X8);

  // QKV projections (pure bf16 WMMA, f32 accumulate)
  proj_kernel<<<1024, 128, 0, stream>>>(xq_bf, wq_bf, bq, 0, q_bf, nullptr);
  proj_kernel<<<1024, 128, 0, stream>>>(xk_bf, wk_bf, bk, 1, k_bf, nullptr);
  proj_kernel<<<1024, 128, 0, stream>>>(xv_bf, wv_bf, bv, 2, v_t,  nullptr);
  // softmax row statistics
  stats_kernel<<<1024, 128, 0, stream>>>(q_bf, k_bf, amask, stats);
  // probabilities, P@V, head-averaged attention weights
  attn_kernel<<<512, 256, 0, stream>>>(q_bf, k_bf, v_t, amask, stats, o_bf, aw);
  // output projection straight into d_out
  proj_kernel<<<1024, 128, 0, stream>>>(o_bf, wo_bf, bo, 3, nullptr, out);
}
